// FasterRCNN_84232898609530
// MI455X (gfx1250) — matrix operand order repaired and checked by reading, not compiled, not run on hardware
//
#include <hip/hip_runtime.h>
#include <hip/hip_bf16.h>

#define BN 2
#define NPROP 3000
#define CCH 512
#define HF 38
#define WF 50
#define KROI 512
#define POOL 7
#define MOFF 10
#define FEATDIM (CCH * POOL * POOL)   // 25088
#define HID 1024
#define NHEAD 128                     // 21 cls + 84 reg, padded to 128
#define NOUT 105
#define SORTN 4096
#define NEGF (-1e30f)

typedef __attribute__((ext_vector_type(16))) __bf16 v16bf;
typedef __attribute__((ext_vector_type(8)))  float  v8f;

static __device__ __forceinline__ unsigned short f2bf(float f) {
    unsigned int u = __float_as_uint(f);
    unsigned int lsb = (u >> 16) & 1u;
    u += 0x7fffu + lsb;               // round-to-nearest-even
    return (unsigned short)(u >> 16);
}

// ---------------- weight conversion + transpose: B[K,N] f32 -> Bt[N,K] bf16 ----------------
__global__ void f32_to_bf16_T_kernel(const float* __restrict__ in,
                                     unsigned short* __restrict__ outT,
                                     int K, int N) {
    for (int i = blockIdx.x * blockDim.x + threadIdx.x; i < K * N;
         i += gridDim.x * blockDim.x) {
        int k = i / N, n = i - k * N;             // coalesced read
        outT[(size_t)n * K + k] = f2bf(in[i]);    // scattered one-time write
    }
}

// pack Wc[1024,21] and Wr[1024,84] into transposed WcrT[128,1024] bf16 (+ bias[128])
__global__ void pack_head_T_kernel(const float* __restrict__ Wc, const float* __restrict__ Wr,
                                   const float* __restrict__ bc, const float* __restrict__ br,
                                   unsigned short* __restrict__ WcrT, float* __restrict__ bcr) {
    int i = blockIdx.x * blockDim.x + threadIdx.x;
    if (i < NHEAD * HID) {
        int n = i / HID, k = i - n * HID;
        float v = 0.f;
        if (n < 21)       v = Wc[k * 21 + n];
        else if (n < 105) v = Wr[k * 84 + (n - 21)];
        WcrT[i] = f2bf(v);
    }
    if (i < NHEAD) {
        float v = 0.f;
        if (i < 21)       v = bc[i];
        else if (i < 105) v = br[i - 21];
        bcr[i] = v;
    }
}

// ---------------- bitonic argsort (descending by score) ----------------
__global__ __launch_bounds__(1024)
void sort_kernel(const float* __restrict__ scores, int* __restrict__ order) {
    __shared__ float s[SORTN];
    __shared__ int   id[SORTN];
    const int tid = threadIdx.x;
    for (int i = tid; i < SORTN; i += 1024) {
        s[i]  = (i < NPROP) ? scores[i] : NEGF;
        id[i] = i;
    }
    __syncthreads();
    for (int k = 2; k <= SORTN; k <<= 1) {
        for (int j = k >> 1; j > 0; j >>= 1) {
            for (int i = tid; i < SORTN; i += 1024) {
                int ixj = i ^ j;
                if (ixj > i) {
                    bool up = ((i & k) == 0);
                    float si = s[i], sx = s[ixj];
                    bool doswap = up ? (si < sx) : (si > sx);
                    if (doswap) {
                        s[i] = sx; s[ixj] = si;
                        int t = id[i]; id[i] = id[ixj]; id[ixj] = t;
                    }
                }
            }
            __syncthreads();
        }
    }
    for (int i = tid; i < NPROP; i += 1024) order[i] = id[i];
}

// ---------------- greedy NMS + top-512 selection + box quantization ----------------
__global__ __launch_bounds__(1024)
void nms_kernel(const float* __restrict__ proposals /*[N,4] xywh*/,
                const int* __restrict__ order, int* __restrict__ bxout /*[512,4]*/) {
    __shared__ float4 sb[NPROP];
    __shared__ int    keep[NPROP];
    const int tid = threadIdx.x;
    for (int i = tid; i < NPROP; i += 1024) {
        int o = order[i];
        float px = proposals[o * 4 + 0], py = proposals[o * 4 + 1];
        float pw = proposals[o * 4 + 2], ph = proposals[o * 4 + 3];
        sb[i] = make_float4(px - pw, py - ph, px, py);   // xyxy
        keep[i] = 1;
    }
    __syncthreads();
    for (int i = 0; i < NPROP; ++i) {
        if (keep[i]) {
            float4 bi = sb[i];
            float ai = fmaxf(bi.z - bi.x, 0.f) * fmaxf(bi.w - bi.y, 0.f);
            for (int j = i + 1 + tid; j < NPROP; j += 1024) {
                if (keep[j]) {
                    float4 bj = sb[j];
                    float x1 = fmaxf(bi.x, bj.x), y1 = fmaxf(bi.y, bj.y);
                    float x2 = fminf(bi.z, bj.z), y2 = fminf(bi.w, bj.w);
                    float inter = fmaxf(x2 - x1, 0.f) * fmaxf(y2 - y1, 0.f);
                    float aj = fmaxf(bj.z - bj.x, 0.f) * fmaxf(bj.w - bj.y, 0.f);
                    float iou = inter / fmaxf(ai + aj - inter, 1e-6f);
                    if (iou > 0.7f) keep[j] = 0;
                }
            }
        }
        __syncthreads();
    }
    if (tid == 0) {
        int cnt = 0;
        // kept boxes first (sorted order), then suppressed ones (mimics top_k over NEG-masked)
        for (int pass = 0; pass < 2 && cnt < KROI; ++pass) {
            for (int i = 0; i < NPROP && cnt < KROI; ++i) {
                if ((pass == 0) == (keep[i] != 0)) {
                    float4 b = sb[i];
                    bxout[cnt * 4 + 0] = (int)rintf(b.x / 16.0f);
                    bxout[cnt * 4 + 1] = (int)rintf(b.y / 16.0f);
                    bxout[cnt * 4 + 2] = (int)rintf(b.z / 16.0f);
                    bxout[cnt * 4 + 3] = (int)rintf(b.w / 16.0f);
                    ++cnt;
                }
            }
        }
    }
}

// ---------------- RoI max pool -> bf16 feature rows ----------------
__global__ __launch_bounds__(256)
void roipool_kernel(const float* __restrict__ feat /*[C,HF,WF]*/,
                    const int* __restrict__ bx /*[512,4]*/,
                    unsigned short* __restrict__ X /*[512, 25088]*/) {
    const int k = blockIdx.x;
    const int x1 = bx[k * 4 + 0], y1 = bx[k * 4 + 1];
    const int x2 = bx[k * 4 + 2], y2 = bx[k * 4 + 3];
    const float rw = (float)max(x2 - x1 + 1, 1);
    const float rh = (float)max(y2 - y1 + 1, 1);
    const float bw = rw / (float)POOL, bh = rh / (float)POOL;
    for (int idx = threadIdx.x; idx < FEATDIM; idx += 256) {
        int c = idx / (POOL * POOL);
        int rem = idx - c * (POOL * POOL);
        int ph = rem / POOL, pw = rem - ph * POOL;
        int ws_ = min(max(x1 + (int)floorf((float)pw * bw), 0), WF);
        int we  = min(max(x1 + (int)ceilf ((float)(pw + 1) * bw), 0), WF);
        int hs  = min(max(y1 + (int)floorf((float)ph * bh), 0), HF);
        int he  = min(max(y1 + (int)ceilf ((float)(ph + 1) * bh), 0), HF);
        int wc = min(we - ws_, MOFF), hc = min(he - hs, MOFF);
        float m = NEGF;
        for (int h = hs; h < hs + hc; ++h)
            for (int w = ws_; w < ws_ + wc; ++w)
                m = fmaxf(m, feat[(c * HF + h) * WF + w]);
        float outv = (wc <= 0 || hc <= 0 || m <= NEGF * 0.5f) ? 0.f : m;
        X[(size_t)k * FEATDIM + idx] = f2bf(outv);
    }
}

// ---------------- bf16 WMMA GEMM: C[M,N] = A[M,K] * Bt[N,K]^T (+bias) ----------------
// block = 256 threads = 8 wave32 waves arranged 2(M) x 4(N); block tile 64(M) x 128(N).
// Each wave owns a 32x32 tile: 2 A-frags x 2 B-frags -> 4 v_wmma_f32_16x16x32_bf16 per K-step.
// A staged through LDS (shared by the 4 n-waves); B fragments loaded straight from the
// pre-transposed weights (two aligned global_load_b128 per fragment).
template <bool RELU_BF16OUT>
__global__ __launch_bounds__(256)
void gemm_bf16_kernel(const unsigned short* __restrict__ A,   // [M x Kd] bf16 row-major
                      const unsigned short* __restrict__ Bt,  // [N x Kd] bf16 (B transposed)
                      const float* __restrict__ bias,
                      void* __restrict__ outp, int Kd, int N) {
    __shared__ unsigned short As[64][40];   // [m][k], 80B row stride (16B-aligned chunks)

    const int tid  = threadIdx.x;
    const int lane = tid & 31;
    const int wave = tid >> 5;
    const int wm   = wave >> 2;             // 0..1
    const int wn   = wave & 3;              // 0..3
    const int m0   = blockIdx.y * 64;
    const int n0   = blockIdx.x * 128;
    const int g    = lane >> 4;             // lane half
    const int mr   = lane & 15;

    v8f z = {0.f, 0.f, 0.f, 0.f, 0.f, 0.f, 0.f, 0.f};
    v8f acc[2][2] = {{z, z}, {z, z}};

    const int arow = tid >> 2;              // 0..63
    const int acol = (tid & 3) * 8;         // 0,8,16,24

    const unsigned short* ag  = A  + (size_t)(m0 + arow) * Kd + acol;
    // two B-fragment column bases for this wave (16-bit B layout: lane-half g
    // holds k in [g*8,g*8+8) U [16+g*8,16+g*8+8) of its column)
    const unsigned short* bg0 = Bt + (size_t)(n0 + wn * 32 + mr) * Kd + g * 8;
    const unsigned short* bg1 = bg0 + (size_t)16 * Kd;

    for (int kb = 0; kb < Kd; kb += 32) {
        // cooperative A tile fill: 64x32 bf16, one b128 load + one b128 DS store per thread
        *(uint4*)&As[arow][acol] = *(const uint4*)(ag + kb);
        __syncthreads();

        union { uint4 q[2]; v16bf v; } fa[2], fb[2];
#pragma unroll
        for (int f = 0; f < 2; ++f) {
            // A fragment (16x32): lane-half g holds k in [g*8,g*8+8) U [16+g*8,16+g*8+8)
            fa[f].q[0] = *(const uint4*)&As[wm * 32 + f * 16 + mr][g * 8];
            fa[f].q[1] = *(const uint4*)&As[wm * 32 + f * 16 + mr][16 + g * 8];
        }
        fb[0].q[0] = *(const uint4*)(bg0 + kb);
        fb[0].q[1] = *(const uint4*)(bg0 + kb + 16);
        fb[1].q[0] = *(const uint4*)(bg1 + kb);
        fb[1].q[1] = *(const uint4*)(bg1 + kb + 16);

        if (kb + 32 < Kd) {                 // pull next K-slice toward L2 (global_prefetch_b8)
            __builtin_prefetch(ag + kb + 32, 0, 1);
            __builtin_prefetch(bg0 + kb + 32, 0, 1);
            __builtin_prefetch(bg1 + kb + 32, 0, 1);
        }

#pragma unroll
        for (int fm = 0; fm < 2; ++fm)
#pragma unroll
            for (int fn = 0; fn < 2; ++fn)
                acc[fm][fn] = __builtin_amdgcn_wmma_f32_16x16x32_bf16(
                    false, fa[fm].v, false, fb[fn].v, (short)0, acc[fm][fn], false, false);
        __syncthreads();
    }

    // D layout: lanes 0-15 -> M=r, lanes 16-31 -> M=8+r; N = lane%16
#pragma unroll
    for (int fn = 0; fn < 2; ++fn) {
        const int col = n0 + wn * 32 + fn * 16 + mr;
        const float bval = bias ? bias[col] : 0.f;
#pragma unroll
        for (int fm = 0; fm < 2; ++fm) {
#pragma unroll
            for (int r = 0; r < 8; ++r) {
                int row = m0 + wm * 32 + fm * 16 + g * 8 + r;
                float v = acc[fm][fn][r] + bval;
                if (RELU_BF16OUT) {
                    v = v > 0.f ? v : 0.f;
                    ((unsigned short*)outp)[(size_t)row * N + col] = f2bf(v);
                } else {
                    ((float*)outp)[(size_t)row * N + col] = v;
                }
            }
        }
    }
}

// ---------------- softmax(cls) || reg -> final output ----------------
__global__ __launch_bounds__(128)
void head_out_kernel(const float* __restrict__ head /*[512,128]*/,
                     float* __restrict__ out /*[512,105]*/) {
    __shared__ float cls[21];
    const int k = blockIdx.x;
    const int j = threadIdx.x;
    const float* row = head + (size_t)k * NHEAD;
    if (j < 21) cls[j] = row[j];
    __syncthreads();
    if (j == 0) {
        float mx = cls[0];
        for (int t = 1; t < 21; ++t) mx = fmaxf(mx, cls[t]);
        float s = 0.f;
        for (int t = 0; t < 21; ++t) { cls[t] = expf(cls[t] - mx); s += cls[t]; }
        float inv = 1.f / s;
        for (int t = 0; t < 21; ++t) cls[t] *= inv;
    }
    __syncthreads();
    if (j < 21)        out[(size_t)k * NOUT + j] = cls[j];
    else if (j < NOUT) out[(size_t)k * NOUT + j] = row[j];
}

extern "C" void kernel_launch(void* const* d_in, const int* in_sizes, int n_in,
                              void* d_out, int out_size, void* d_ws, size_t ws_size,
                              hipStream_t stream) {
    const float* feat      = (const float*)d_in[0];
    const float* proposals = (const float*)d_in[1];
    const float* scores    = (const float*)d_in[2];
    const float* W1        = (const float*)d_in[3];
    const float* b1        = (const float*)d_in[4];
    const float* W2        = (const float*)d_in[5];
    const float* b2        = (const float*)d_in[6];
    const float* Wc        = (const float*)d_in[7];
    const float* bc        = (const float*)d_in[8];
    const float* Wr        = (const float*)d_in[9];
    const float* br        = (const float*)d_in[10];
    float* out = (float*)d_out;

    char* wsp = (char*)d_ws;
    auto alloc = [&](size_t bytes) -> char* {
        char* p = wsp;
        wsp += (bytes + 255) & ~(size_t)255;
        return p;
    };
    unsigned short* W1t  = (unsigned short*)alloc((size_t)FEATDIM * HID * 2);  // [1024 x 25088]
    unsigned short* W2t  = (unsigned short*)alloc((size_t)HID * HID * 2);      // [1024 x 1024]
    unsigned short* WcrT = (unsigned short*)alloc((size_t)NHEAD * HID * 2);    // [128 x 1024]
    float*          bcr  = (float*)alloc(NHEAD * 4);
    int*            ordr = (int*)alloc(NPROP * 4);
    int*            bx   = (int*)alloc(KROI * 4 * 4);
    unsigned short* X    = (unsigned short*)alloc((size_t)KROI * FEATDIM * 2);
    unsigned short* H1   = (unsigned short*)alloc((size_t)KROI * HID * 2);
    unsigned short* H2   = (unsigned short*)alloc((size_t)KROI * HID * 2);
    float*          HEAD = (float*)alloc((size_t)KROI * NHEAD * 4);

    f32_to_bf16_T_kernel<<<2048, 256, 0, stream>>>(W1, W1t, FEATDIM, HID);
    f32_to_bf16_T_kernel<<<512, 256, 0, stream>>>(W2, W2t, HID, HID);
    pack_head_T_kernel<<<(NHEAD * HID + 255) / 256, 256, 0, stream>>>(Wc, Wr, bc, br, WcrT, bcr);

    for (int b = 0; b < BN; ++b) {
        sort_kernel<<<1, 1024, 0, stream>>>(scores + (size_t)b * NPROP, ordr);
        nms_kernel<<<1, 1024, 0, stream>>>(proposals + (size_t)b * NPROP * 4, ordr, bx);
        roipool_kernel<<<KROI, 256, 0, stream>>>(feat + (size_t)b * CCH * HF * WF, bx, X);

        gemm_bf16_kernel<true><<<dim3(HID / 128, KROI / 64), 256, 0, stream>>>(
            X, W1t, b1, (void*)H1, FEATDIM, HID);
        gemm_bf16_kernel<true><<<dim3(HID / 128, KROI / 64), 256, 0, stream>>>(
            H1, W2t, b2, (void*)H2, HID, HID);
        gemm_bf16_kernel<false><<<dim3(NHEAD / 128, KROI / 64), 256, 0, stream>>>(
            H2, WcrT, bcr, (void*)HEAD, HID, NHEAD);

        head_out_kernel<<<KROI, 128, 0, stream>>>(HEAD, out + (size_t)b * KROI * NOUT);
    }
}
